// _Transformer_58729382805707
// MI455X (gfx1250) — matrix operand order repaired and checked by reading, not compiled, not run on hardware
//
#include <hip/hip_runtime.h>
#include <hip/hip_bf16.h>
#include <math.h>

// ---------------- model constants ----------------
#define BSZ    2
#define SEQ    1024
#define DM     1024
#define NH     16
#define HD     64
#define DEPTH  4
#define INNER  2730
#define INNERP 2752            // INNER padded to multiple of 32 (16B-aligned rows)
#define FF1N   (2*INNER)       // 5460
#define TD     (3*DM)          // 3072
#define NTOK   (BSZ*SEQ)       // 2048
#define RMS_SCALE 32.0f        // sqrt(DM)

typedef __attribute__((ext_vector_type(16))) __bf16 v16bf;
typedef __attribute__((ext_vector_type(8)))  float  v8f;

// ---------------------------------------------------------------------
// CDNA5 async global->LDS copy (ASYNCcnt-tracked), with sync fallback.
// Builtin signature (probed via diagnostics):
//   void __builtin_amdgcn_global_load_async_to_lds_b128(
//        v4i32 addrspace(1)* src, v4i32 addrspace(3)* dst, imm offset, imm cpol)
// ---------------------------------------------------------------------
#if __has_builtin(__builtin_amdgcn_global_load_async_to_lds_b128)
#define HAVE_ASYNC_LDS 1
#else
#define HAVE_ASYNC_LDS 0
#endif

typedef int i32x4 __attribute__((vector_size(16)));
typedef __attribute__((address_space(1))) i32x4 gas_i32x4;
typedef __attribute__((address_space(3))) i32x4 las_i32x4;

__device__ __forceinline__ void async_b128(const __bf16* g, __bf16* l) {
#if HAVE_ASYNC_LDS
    __builtin_amdgcn_global_load_async_to_lds_b128(
        (gas_i32x4*)(void*)const_cast<__bf16*>(g),
        (las_i32x4*)(void*)l, 0, 0);
#else
    *(uint4*)l = *(const uint4*)g;
#endif
}

__device__ __forceinline__ void wait_async_0() {
#if HAVE_ASYNC_LDS
#if __has_builtin(__builtin_amdgcn_s_wait_asynccnt)
    __builtin_amdgcn_s_wait_asynccnt(0);
#else
    asm volatile("s_wait_asynccnt 0x0" ::: "memory");
#endif
#endif
}

__device__ __forceinline__ void wait_async_3() {
#if HAVE_ASYNC_LDS
#if __has_builtin(__builtin_amdgcn_s_wait_asynccnt)
    __builtin_amdgcn_s_wait_asynccnt(3);
#else
    asm volatile("s_wait_asynccnt 0x3" ::: "memory");
#endif
#endif
}

// =====================================================================
// Elementwise / conversion kernels
// =====================================================================
__global__ void k_f32_to_bf16(const float* __restrict__ s, __bf16* __restrict__ d, long n) {
    long i = (long)blockIdx.x * blockDim.x + threadIdx.x;
    long stride = (long)gridDim.x * blockDim.x;
    for (; i < n; i += stride) d[i] = (__bf16)s[i];
}

__global__ void k_cvt_pad(const float* __restrict__ s, __bf16* __restrict__ d,
                          int rows, int K, int Kp) {
    long n = (long)rows * Kp;
    long i = (long)blockIdx.x * blockDim.x + threadIdx.x;
    long stride = (long)gridDim.x * blockDim.x;
    for (; i < n; i += stride) {
        int row = (int)(i / Kp);
        int kp  = (int)(i % Kp);
        d[i] = (kp < K) ? (__bf16)s[(long)row * K + kp] : (__bf16)0.0f;
    }
}

__global__ void k_rmsnorm_bf16(const float* __restrict__ x, const float* __restrict__ gamma,
                               __bf16* __restrict__ out) {
    __shared__ float red[256];
    const int row = blockIdx.x;
    const float* xr = x + (long)row * DM;
    float ss = 0.f;
    for (int c = threadIdx.x; c < DM; c += 256) { float v = xr[c]; ss += v * v; }
    red[threadIdx.x] = ss;
    __syncthreads();
    for (int off = 128; off > 0; off >>= 1) {
        if (threadIdx.x < off) red[threadIdx.x] += red[threadIdx.x + off];
        __syncthreads();
    }
    const float sc = RMS_SCALE / fmaxf(sqrtf(red[0]), 1e-12f);
    for (int c = threadIdx.x; c < DM; c += 256)
        out[(long)row * DM + c] = (__bf16)(xr[c] * sc * gamma[c]);
}

__global__ void k_rmsnorm_f32(const float* __restrict__ x, const float* __restrict__ gamma,
                              float* __restrict__ out) {
    __shared__ float red[256];
    const int row = blockIdx.x;
    const float* xr = x + (long)row * DM;
    float ss = 0.f;
    for (int c = threadIdx.x; c < DM; c += 256) { float v = xr[c]; ss += v * v; }
    red[threadIdx.x] = ss;
    __syncthreads();
    for (int off = 128; off > 0; off >>= 1) {
        if (threadIdx.x < off) red[threadIdx.x] += red[threadIdx.x + off];
        __syncthreads();
    }
    const float sc = RMS_SCALE / fmaxf(sqrtf(red[0]), 1e-12f);
    for (int c = threadIdx.x; c < DM; c += 256)
        out[(long)row * DM + c] = xr[c] * sc * gamma[c];
}

// qkv (B*L, 3D) f32 -> q/k (B,H,L,HD) bf16, v -> V^T (B,H,HD,L) bf16
__global__ void k_qkv_split(const float* __restrict__ qkv,
                            __bf16* __restrict__ q, __bf16* __restrict__ k,
                            __bf16* __restrict__ vt) {
    int i = blockIdx.x * 256 + threadIdx.x;          // over B*L*D
    if (i >= NTOK * DM) return;
    int row = i >> 10;                                // b*SEQ + l
    int col = i & (DM - 1);
    int h = col >> 6, dd = col & (HD - 1);
    int b = row >> 10, l = row & (SEQ - 1);
    long src = (long)row * TD;
    long bh = (long)(b * NH + h);
    long dst = (bh * SEQ + l) * HD + dd;
    q[dst] = (__bf16)qkv[src + col];
    k[dst] = (__bf16)qkv[src + DM + col];
    vt[(bh * HD + dd) * SEQ + l] = (__bf16)qkv[src + 2 * DM + col];
}

__global__ void k_geglu(const float* __restrict__ h, __bf16* __restrict__ g) {
    long n = (long)NTOK * INNERP;
    long i = (long)blockIdx.x * 256 + threadIdx.x;
    if (i >= n) return;
    int row = (int)(i / INNERP);
    int col = (int)(i % INNERP);
    float o = 0.f;
    if (col < INNER) {
        float a  = h[(long)row * FF1N + col];
        float gt = h[(long)row * FF1N + INNER + col];
        float ge = 0.5f * gt * (1.0f + erff(gt * 0.70710678118654752f));
        o = ge * a;
    }
    g[i] = (__bf16)o;
}

__global__ void k_softmax(const float* __restrict__ S, __bf16* __restrict__ P) {
    __shared__ float red[256];
    const long row = blockIdx.x;
    const float* sr = S + row * SEQ;
    __bf16* pr = P + row * SEQ;
    float mx = -1e30f;
    float vals[4];
    for (int k = 0; k < 4; k++) { vals[k] = sr[threadIdx.x + k * 256]; mx = fmaxf(mx, vals[k]); }
    red[threadIdx.x] = mx;
    __syncthreads();
    for (int off = 128; off > 0; off >>= 1) {
        if (threadIdx.x < off) red[threadIdx.x] = fmaxf(red[threadIdx.x], red[threadIdx.x + off]);
        __syncthreads();
    }
    const float rowmax = red[0];
    __syncthreads();
    float e[4], sum = 0.f;
    for (int k = 0; k < 4; k++) { e[k] = __expf(vals[k] - rowmax); sum += e[k]; }
    red[threadIdx.x] = sum;
    __syncthreads();
    for (int off = 128; off > 0; off >>= 1) {
        if (threadIdx.x < off) red[threadIdx.x] += red[threadIdx.x + off];
        __syncthreads();
    }
    const float inv = 1.0f / red[0];
    for (int k = 0; k < 4; k++) pr[threadIdx.x + k * 256] = (__bf16)(e[k] * inv);
}

// =====================================================================
// WMMA fragment loaders (CDNA5 ISA §7.12.2, wave32) — b128 LDS loads.
//   A 16x32: lane l (m=l&15): elems 0..7 = K[kh..kh+7], 8..15 = K[kh+16..kh+23], kh=(l>>4)*8
//   B 32x16: lane l (n=l&15): elems 0..7 = K[kh..kh+7], 8..15 = K[kh+16..kh+23], kh=(l>>4)*8
//   C 16x16: VGPR r of lane l -> m=(l>>4)*8+r, n=l&15
// =====================================================================
__device__ __forceinline__ v16bf load_a_frag(const __bf16* rowp, int lane) {
    const int kh = (lane >> 4) * 8;
    union { uint4 q[2]; v16bf v; } u;
    u.q[0] = *(const uint4*)(rowp + kh);
    u.q[1] = *(const uint4*)(rowp + kh + 16);
    return u.v;
}
__device__ __forceinline__ v16bf load_b_frag(const __bf16* rowp, int lane) {
    const int kh = (lane >> 4) * 8;
    union { uint4 q[2]; v16bf v; } u;
    u.q[0] = *(const uint4*)(rowp + kh);
    u.q[1] = *(const uint4*)(rowp + kh + 16);
    return u.v;
}

#define LDS_S 40   // row stride (bf16 elems) = 80B, multiple of 16B

// =====================================================================
// Generic GEMM: C[M,N] = A[M,K](bf16) @ W[N,K]^T(bf16) + bias[N] (+res)
// 256 thr = 8 waves; 128x64 tile; double-buffered async LDS staging.
// =====================================================================
__global__ __launch_bounds__(256)
void k_gemm(const __bf16* __restrict__ A, const __bf16* __restrict__ W,
            const float* __restrict__ bias, const float* __restrict__ res,
            float* __restrict__ C, int M, int N, int K) {
    __shared__ __align__(16) __bf16 As[2][128][LDS_S];
    __shared__ __align__(16) __bf16 Bs[2][64][LDS_S];
    const int tid = threadIdx.x;
    const int wave = tid >> 5, lane = tid & 31;
    const int m0 = blockIdx.y * 128, n0 = blockIdx.x * 64;

    auto issue = [&](int buf, int k0) {
#pragma unroll
        for (int c = 0; c < 2; c++) {                     // A tile: 128x32
            int chunk = tid + c * 256;
            int r = chunk >> 2, col = (chunk & 3) * 8;
            int gm = m0 + r; gm = (gm < M) ? gm : (M - 1);   // clamp: keep EXEC uniform
            async_b128(A + (long)gm * K + k0 + col, &As[buf][r][col]);
        }
        {                                                 // B tile: 64x32
            int r = tid >> 2, col = (tid & 3) * 8;
            int gn = n0 + r; gn = (gn < N) ? gn : (N - 1);
            async_b128(W + (long)gn * K + k0 + col, &Bs[buf][r][col]);
        }
    };

    v8f acc[4] = {};
    issue(0, 0);
    const int steps = K >> 5;
    for (int s = 0; s < steps; s++) {
        const int buf = s & 1;
        if (s + 1 < steps) { issue(buf ^ 1, (s + 1) << 5); wait_async_3(); }
        else               { wait_async_0(); }
        __syncthreads();
        v16bf a = load_a_frag(&As[buf][wave * 16 + (lane & 15)][0], lane);
#pragma unroll
        for (int j = 0; j < 4; j++) {
            v16bf b = load_b_frag(&Bs[buf][j * 16 + (lane & 15)][0], lane);
            acc[j] = __builtin_amdgcn_wmma_f32_16x16x32_bf16(
                false, a, false, b, (short)0, acc[j], false, false);
        }
        __syncthreads();
    }
#pragma unroll
    for (int j = 0; j < 4; j++) {
#pragma unroll
        for (int r = 0; r < 8; r++) {
            int gm = m0 + wave * 16 + (lane >> 4) * 8 + r;
            int gn = n0 + j * 16 + (lane & 15);
            if (gm < M && gn < N) {
                float v = acc[j][r] + bias[gn];
                if (res) v += res[(long)gm * N + gn];
                C[(long)gm * N + gn] = v;
            }
        }
    }
}

// =====================================================================
// Attention scores: S[b,h,q,k] = (Q.K^T)/8 - |q-k|*slope[h]
// 128 thr = 4 waves; 64x64 tile; K-dim = HD = 64 staged once (async).
// =====================================================================
__global__ __launch_bounds__(128)
void k_attn_scores(const __bf16* __restrict__ Q, const __bf16* __restrict__ Kt,
                   const float* __restrict__ slopes, float* __restrict__ S) {
    __shared__ __align__(16) __bf16 Qs[64][72];
    __shared__ __align__(16) __bf16 Ks[64][72];
    const int bh = blockIdx.z;
    const int h = bh & (NH - 1);
    const __bf16* Qb = Q + (long)bh * SEQ * HD;
    const __bf16* Kb = Kt + (long)bh * SEQ * HD;
    float* Sb = S + (long)bh * SEQ * SEQ;
    const int tid = threadIdx.x, wave = tid >> 5, lane = tid & 31;
    const int m0 = blockIdx.y * 64, n0 = blockIdx.x * 64;

#pragma unroll
    for (int c = 0; c < 4; c++) {                       // 64x64 each
        int chunk = tid + c * 128;
        int r = chunk >> 3, col = (chunk & 7) * 8;
        async_b128(Qb + (long)(m0 + r) * HD + col, &Qs[r][col]);
        async_b128(Kb + (long)(n0 + r) * HD + col, &Ks[r][col]);
    }
    wait_async_0();
    __syncthreads();

    v8f acc[4] = {};
#pragma unroll
    for (int k0 = 0; k0 < HD; k0 += 32) {
        v16bf a = load_a_frag(&Qs[wave * 16 + (lane & 15)][k0], lane);
#pragma unroll
        for (int j = 0; j < 4; j++) {
            v16bf b = load_b_frag(&Ks[j * 16 + (lane & 15)][k0], lane);
            acc[j] = __builtin_amdgcn_wmma_f32_16x16x32_bf16(
                false, a, false, b, (short)0, acc[j], false, false);
        }
    }
    const float slope = slopes[h];
#pragma unroll
    for (int j = 0; j < 4; j++) {
#pragma unroll
        for (int r = 0; r < 8; r++) {
            int q = m0 + wave * 16 + (lane >> 4) * 8 + r;
            int k = n0 + j * 16 + (lane & 15);
            Sb[(long)q * SEQ + k] = acc[j][r] * 0.125f - fabsf((float)(q - k)) * slope;
        }
    }
}

// =====================================================================
// O = P @ V  using V^T (B,H,HD,SEQ): per (b,h), M=SEQ, N=HD, K=SEQ.
// Same structure as k_gemm; writes bf16 straight into (B,L,D).
// =====================================================================
__global__ __launch_bounds__(256)
void k_attn_pv(const __bf16* __restrict__ P, const __bf16* __restrict__ Vt,
               __bf16* __restrict__ O) {
    __shared__ __align__(16) __bf16 As[2][128][LDS_S];
    __shared__ __align__(16) __bf16 Bs[2][64][LDS_S];
    const int bh = blockIdx.z;
    const int b = bh >> 4, h = bh & (NH - 1);
    const __bf16* Pb = P + (long)bh * SEQ * SEQ;
    const __bf16* Wb = Vt + (long)bh * HD * SEQ;       // (N=HD rows, K=SEQ)
    const int tid = threadIdx.x, wave = tid >> 5, lane = tid & 31;
    const int m0 = blockIdx.y * 128;

    auto issue = [&](int buf, int k0) {
#pragma unroll
        for (int c = 0; c < 2; c++) {
            int chunk = tid + c * 256;
            int r = chunk >> 2, col = (chunk & 3) * 8;
            async_b128(Pb + (long)(m0 + r) * SEQ + k0 + col, &As[buf][r][col]);
        }
        {
            int r = tid >> 2, col = (tid & 3) * 8;
            async_b128(Wb + (long)r * SEQ + k0 + col, &Bs[buf][r][col]);
        }
    };

    v8f acc[4] = {};
    issue(0, 0);
    const int steps = SEQ >> 5;
    for (int s = 0; s < steps; s++) {
        const int buf = s & 1;
        if (s + 1 < steps) { issue(buf ^ 1, (s + 1) << 5); wait_async_3(); }
        else               { wait_async_0(); }
        __syncthreads();
        v16bf a = load_a_frag(&As[buf][wave * 16 + (lane & 15)][0], lane);
#pragma unroll
        for (int j = 0; j < 4; j++) {
            v16bf bf = load_b_frag(&Bs[buf][j * 16 + (lane & 15)][0], lane);
            acc[j] = __builtin_amdgcn_wmma_f32_16x16x32_bf16(
                false, a, false, bf, (short)0, acc[j], false, false);
        }
        __syncthreads();
    }
#pragma unroll
    for (int j = 0; j < 4; j++) {
#pragma unroll
        for (int r = 0; r < 8; r++) {
            int q = m0 + wave * 16 + (lane >> 4) * 8 + r;
            int n = j * 16 + (lane & 15);
            O[((long)(b * SEQ + q)) * DM + h * HD + n] = (__bf16)acc[j][r];
        }
    }
}

// =====================================================================
// Host-side launcher
// =====================================================================
static inline size_t rup(size_t x) { return (x + 255) & ~(size_t)255; }

extern "C" void kernel_launch(void* const* d_in, const int* in_sizes, int n_in,
                              void* d_out, int out_size, void* d_ws, size_t ws_size,
                              hipStream_t stream) {
    (void)in_sizes; (void)n_in; (void)out_size; (void)ws_size;
    const float* x_in    = (const float*)d_in[0];
    const float* slopes  = (const float*)d_in[1];
    const float* g_attn  = (const float*)d_in[2];
    const float* in_w    = (const float*)d_in[3];
    const float* in_b    = (const float*)d_in[4];
    const float* out_w   = (const float*)d_in[5];
    const float* out_b   = (const float*)d_in[6];
    const float* g_ff    = (const float*)d_in[7];
    const float* ff_w1   = (const float*)d_in[8];
    const float* ff_b1   = (const float*)d_in[9];
    const float* ff_w2   = (const float*)d_in[10];
    const float* ff_b2   = (const float*)d_in[11];
    const float* g_final = (const float*)d_in[12];

    char* base = (char*)d_ws;
    size_t off = 0;
    float*  xbuf   = (float*)(base + off);  off += rup((size_t)NTOK * DM * 4);
    __bf16* nrm    = (__bf16*)(base + off); off += rup((size_t)NTOK * DM * 2);
    float*  qkvf   = (float*)(base + off);  off += rup((size_t)NTOK * TD * 4);
    __bf16* qb     = (__bf16*)(base + off); off += rup((size_t)NTOK * DM * 2);
    __bf16* kb     = (__bf16*)(base + off); off += rup((size_t)NTOK * DM * 2);
    __bf16* vtb    = (__bf16*)(base + off); off += rup((size_t)NTOK * DM * 2);
    __bf16* attn_o = (__bf16*)(base + off); off += rup((size_t)NTOK * DM * 2);
    size_t big = off;
    float*  scores = (float*)(base + big);
    __bf16* probs  = (__bf16*)(base + big + rup((size_t)BSZ * NH * SEQ * SEQ * 4));
    float*  ffh    = (float*)(base + big);
    __bf16* geglu  = (__bf16*)(base + big + rup((size_t)NTOK * FF1N * 4));
    off = big + rup((size_t)BSZ * NH * SEQ * SEQ * 4) + rup((size_t)BSZ * NH * SEQ * SEQ * 2);
    __bf16* w_in   = (__bf16*)(base + off); off += rup((size_t)DEPTH * TD * DM * 2);
    __bf16* w_out  = (__bf16*)(base + off); off += rup((size_t)DEPTH * DM * DM * 2);
    __bf16* w_f1   = (__bf16*)(base + off); off += rup((size_t)DEPTH * FF1N * DM * 2);
    __bf16* w_f2   = (__bf16*)(base + off); off += rup((size_t)DEPTH * DM * INNERP * 2);

    {
        long n1 = (long)DEPTH * TD * DM;
        k_f32_to_bf16<<<(int)((n1 + 255) / 256), 256, 0, stream>>>(in_w, w_in, n1);
        long n2 = (long)DEPTH * DM * DM;
        k_f32_to_bf16<<<(int)((n2 + 255) / 256), 256, 0, stream>>>(out_w, w_out, n2);
        long n3 = (long)DEPTH * FF1N * DM;
        k_f32_to_bf16<<<(int)((n3 + 255) / 256), 256, 0, stream>>>(ff_w1, w_f1, n3);
        long n4 = (long)DEPTH * DM * INNERP;
        k_cvt_pad<<<(int)((n4 + 255) / 256), 256, 0, stream>>>(ff_w2, w_f2, DEPTH * DM, INNER, INNERP);
    }
    (void)hipMemcpyAsync(xbuf, x_in, (size_t)NTOK * DM * 4, hipMemcpyDeviceToDevice, stream);

    for (int i = 0; i < DEPTH; i++) {
        k_rmsnorm_bf16<<<NTOK, 256, 0, stream>>>(xbuf, g_attn + (size_t)i * DM, nrm);
        k_gemm<<<dim3(TD / 64, NTOK / 128), 256, 0, stream>>>(
            nrm, w_in + (size_t)i * TD * DM, in_b + (size_t)i * TD, nullptr,
            qkvf, NTOK, TD, DM);
        k_qkv_split<<<(NTOK * DM) / 256, 256, 0, stream>>>(qkvf, qb, kb, vtb);
        k_attn_scores<<<dim3(SEQ / 64, SEQ / 64, BSZ * NH), 128, 0, stream>>>(
            qb, kb, slopes, scores);
        k_softmax<<<BSZ * NH * SEQ, 256, 0, stream>>>(scores, probs);
        k_attn_pv<<<dim3(1, SEQ / 128, BSZ * NH), 256, 0, stream>>>(probs, vtb, attn_o);
        k_gemm<<<dim3(DM / 64, NTOK / 128), 256, 0, stream>>>(
            attn_o, w_out + (size_t)i * DM * DM, out_b + (size_t)i * DM, xbuf,
            xbuf, NTOK, DM, DM);
        k_rmsnorm_bf16<<<NTOK, 256, 0, stream>>>(xbuf, g_ff + (size_t)i * DM, nrm);
        k_gemm<<<dim3((FF1N + 63) / 64, NTOK / 128), 256, 0, stream>>>(
            nrm, w_f1 + (size_t)i * FF1N * DM, ff_b1 + (size_t)i * FF1N, nullptr,
            ffh, NTOK, FF1N, DM);
        {
            long n = (long)NTOK * INNERP;
            k_geglu<<<(int)((n + 255) / 256), 256, 0, stream>>>(ffh, geglu);
        }
        k_gemm<<<dim3(DM / 64, NTOK / 128), 256, 0, stream>>>(
            geglu, w_f2 + (size_t)i * DM * INNERP, ff_b2 + (size_t)i * DM, xbuf,
            xbuf, NTOK, DM, INNERP);
    }
    k_rmsnorm_f32<<<NTOK, 256, 0, stream>>>(xbuf, g_final, (float*)d_out);
}
